// TransformerBlock_54778012893611
// MI455X (gfx1250) — matrix operand order repaired and checked by reading, not compiled, not run on hardware
//
#include <hip/hip_runtime.h>
#include <hip/hip_bf16.h>

// ---------------------------------------------------------------------------
// PointTransformerConv fused pipeline for MI455X (gfx1250), wave32 + WMMA bf16
// ---------------------------------------------------------------------------

typedef __attribute__((ext_vector_type(16))) __bf16 bf16x16;
typedef __attribute__((ext_vector_type(8)))  float  f32x8;

#define CDIM 64

__device__ __forceinline__ f32x8 wmma_bf16(bf16x16 a, bf16x16 b, f32x8 c) {
    return __builtin_amdgcn_wmma_f32_16x16x32_bf16(false, a, false, b, (short)0, c,
                                                   false, false);
}

// A fragment (16x32 bf16 tile slice) from an LDS 16x64 row-major tile.
// Lane L: row m = L%16, half h = L/16; elems 0..7 = K[koff+8h .. +7],
// elems 8..15 = K[koff+16+8h .. +7]   (ISA 7.12.2, 16-bit A 16x32)
__device__ __forceinline__ bf16x16 load_a_frag(const __bf16* tile, int lane, int koff) {
    int m = lane & 15, h = lane >> 4;
    union { bf16x16 v; uint4 u[2]; } r;
    r.u[0] = *(const uint4*)(tile + m * CDIM + koff + 8 * h);
    r.u[1] = *(const uint4*)(tile + m * CDIM + koff + 16 + 8 * h);
    return r.v;
}

// B fragment (32x16 bf16) from an LDS weight stored transposed: wt[n][k], 64x64.
// Lane L: col n = ntile*16 + L%16, elems 0..7 = K[koff+8h .. +7], elems 8..15 = K[koff+16+8h .. +7].
__device__ __forceinline__ bf16x16 load_b_frag(const __bf16* wt, int lane, int ntile, int koff) {
    int n  = (lane & 15) + ntile * 16;
    int kb = koff + ((lane >> 4) << 3);
    union { bf16x16 v; uint4 u[2]; } r;
    const uint4* p = (const uint4*)(wt + n * CDIM + kb);
    r.u[0] = p[0];
    r.u[1] = p[2];
    return r.v;
}

__device__ __forceinline__ void ds_fence() {
    asm volatile("s_wait_dscnt 0" ::: "memory");
}

// ---------------------------------------------------------------------------
// Kernel 0: convert 8 weight matrices (64x64 f32 row-major [k][n]) to bf16
// transposed [n][k] so each lane's B-fragment is two contiguous 16B runs.
// wbf layout: mat 0=W_in 1=W_src 2=W_dst 3=W_lin 4=Wp2 5=Wa1 6=Wa2 7=W_out
// ---------------------------------------------------------------------------
__global__ void prep_weights(const float* __restrict__ W_in, const float* __restrict__ W_src,
                             const float* __restrict__ W_dst, const float* __restrict__ W_lin,
                             const float* __restrict__ Wp2,  const float* __restrict__ Wa1,
                             const float* __restrict__ Wa2,  const float* __restrict__ W_out,
                             __bf16* __restrict__ wbf) {
    const float* Ws[8] = {W_in, W_src, W_dst, W_lin, Wp2, Wa1, Wa2, W_out};
    int t = threadIdx.x + blockIdx.x * blockDim.x;
    if (t < 8 * 4096) {
        int mat = t >> 12;
        int idx = t & 4095;
        int n = idx >> 6, k = idx & 63;
        wbf[t] = (__bf16)Ws[mat][k * 64 + n];
    }
}

__global__ void init_zero(unsigned* __restrict__ p, size_t n) {
    size_t i = (size_t)blockIdx.x * blockDim.x + threadIdx.x;
    size_t stride = (size_t)gridDim.x * blockDim.x;
    for (; i < n; i += stride) p[i] = 0u;
}

// ---------------------------------------------------------------------------
// Kernel 1: node transform.  h = relu(x@W_in+b_in); a_src = h@W_src;
// a_dst = h@W_dst; v = h@W_lin+b_lin.  One wave = 16 node rows, 32 WMMAs.
// ---------------------------------------------------------------------------
__global__ __launch_bounds__(256) void node_transform(
    const float* __restrict__ x,
    const float* __restrict__ b_in, const float* __restrict__ b_lin,
    const __bf16* __restrict__ wbf,
    float* __restrict__ a_src, float* __restrict__ a_dst, float* __restrict__ vv,
    int N) {
    __shared__ __align__(16) __bf16 sW[4 * 4096];     // W_in, W_src, W_dst, W_lin
    __shared__ float sbin[64], sblin[64];
    __shared__ __align__(16) __bf16 sTile[8][16 * CDIM];

    int tid = threadIdx.x;
    for (int i = tid; i < 4 * 4096; i += 256) sW[i] = wbf[i];
    if (tid < 64) { sbin[tid] = b_in[tid]; sblin[tid] = b_lin[tid]; }
    __syncthreads();

    int wave = tid >> 5, lane = tid & 31;
    int r0 = (blockIdx.x * 8 + wave) * 16;
    if (r0 >= N) return;
    __bf16* tile = sTile[wave];
    int n0 = lane & 15, hh = lane >> 4;

    // Stage x tile (16x64) into LDS as bf16
#pragma unroll
    for (int i = 0; i < 8; ++i) {
        int e4 = (lane + i * 32) * 4;
        int row = e4 >> 6, col = e4 & 63;
        float4 xv = make_float4(0.f, 0.f, 0.f, 0.f);
        if (r0 + row < N) xv = *(const float4*)(x + (size_t)(r0 + row) * CDIM + col);
        tile[row * CDIM + col + 0] = (__bf16)xv.x;
        tile[row * CDIM + col + 1] = (__bf16)xv.y;
        tile[row * CDIM + col + 2] = (__bf16)xv.z;
        tile[row * CDIM + col + 3] = (__bf16)xv.w;
    }
    ds_fence();

    bf16x16 xa0 = load_a_frag(tile, lane, 0);
    bf16x16 xa1 = load_a_frag(tile, lane, 32);

    // h = relu(x @ W_in + b_in), written back to the tile in bf16
#pragma unroll
    for (int t = 0; t < 4; ++t) {
        f32x8 acc = {};
        acc = wmma_bf16(xa0, load_b_frag(sW, lane, t, 0), acc);
        acc = wmma_bf16(xa1, load_b_frag(sW, lane, t, 32), acc);
        float bi = sbin[t * 16 + n0];
#pragma unroll
        for (int e = 0; e < 8; ++e) {
            float hv = fmaxf(acc[e] + bi, 0.f);
            tile[(e + 8 * hh) * CDIM + t * 16 + n0] = (__bf16)hv;
        }
    }
    ds_fence();

    bf16x16 ha0 = load_a_frag(tile, lane, 0);
    bf16x16 ha1 = load_a_frag(tile, lane, 32);

    float* outs[3] = {a_src, a_dst, vv};
#pragma unroll
    for (int mi = 0; mi < 3; ++mi) {
        const __bf16* Wm = sW + (mi + 1) * 4096;
        float* op = outs[mi];
#pragma unroll
        for (int t = 0; t < 4; ++t) {
            f32x8 acc = {};
            acc = wmma_bf16(ha0, load_b_frag(Wm, lane, t, 0), acc);
            acc = wmma_bf16(ha1, load_b_frag(Wm, lane, t, 32), acc);
            float bi = (mi == 2) ? sblin[t * 16 + n0] : 0.f;
#pragma unroll
            for (int e = 0; e < 8; ++e) {
                int row = e + 8 * hh;
                if (r0 + row < N)
                    op[(size_t)(r0 + row) * CDIM + t * 16 + n0] = acc[e] + bi;
            }
        }
    }
}

// ---------------------------------------------------------------------------
// Kernels 2/3: edge pipeline.  One wave = 16 edges.
//   delta = relu(relu(d3@Wp1+bp1)@Wp2+bp2)
//   logit = relu(relu((a_dst[d]-a_src[s]+delta)@Wa1+ba1)@Wa2+ba2)
// PASS 1: segment-max of logits into node_max (ordered-uint encoding).
// PASS 2: w = exp(logit - max[dst]); atomic add denom += w,
//         accum += w * (v[src] + delta)   (delta kept in D-layout VGPRs)
// ---------------------------------------------------------------------------
template <int PASS>
__global__ __launch_bounds__(256) void edge_pass(
    const float* __restrict__ pos,
    const int* __restrict__ esrc, const int* __restrict__ edst,
    const float* __restrict__ a_src, const float* __restrict__ a_dst,
    const float* __restrict__ vv,
    const float* __restrict__ Wp1, const float* __restrict__ bp1,
    const float* __restrict__ bp2, const float* __restrict__ ba1,
    const float* __restrict__ ba2,
    const __bf16* __restrict__ wbf,
    unsigned* __restrict__ node_max, float* __restrict__ denom,
    float* __restrict__ accum, int E) {
    __shared__ __align__(16) __bf16 sWp2[4096], sWa1[4096], sWa2[4096];
    __shared__ float sWp1[64 * 4];            // [j][t], t<3
    __shared__ float sbp1[64], sbp2[64], sba1[64], sba2[64];
    __shared__ __align__(16) __bf16 sTile[8][16 * CDIM];
    __shared__ float sD3[8][16][4];
    __shared__ int sSrc[8][16], sDst[8][16];

    int tid = threadIdx.x;
    for (int i = tid; i < 4096; i += 256) {
        sWp2[i] = wbf[4 * 4096 + i];
        sWa1[i] = wbf[5 * 4096 + i];
        sWa2[i] = wbf[6 * 4096 + i];
    }
    if (tid < 64) {
        sbp1[tid] = bp1[tid]; sbp2[tid] = bp2[tid];
        sba1[tid] = ba1[tid]; sba2[tid] = ba2[tid];
        sWp1[tid * 4 + 0] = Wp1[tid];
        sWp1[tid * 4 + 1] = Wp1[64 + tid];
        sWp1[tid * 4 + 2] = Wp1[128 + tid];
        sWp1[tid * 4 + 3] = 0.f;
    }
    __syncthreads();

    int wave = tid >> 5, lane = tid & 31;
    long long e0 = ((long long)blockIdx.x * 8 + wave) * 16;
    if (e0 >= E) return;
    __bf16* tile = sTile[wave];
    int m = lane & 15, hh = lane >> 4, n0 = lane & 15;

    if (lane < 16) {
        long long e = e0 + lane;
        if (e >= E) e = E - 1;
        int s = esrc[e], d = edst[e];
        sSrc[wave][lane] = s;
        sDst[wave][lane] = d;
        sD3[wave][lane][0] = pos[(size_t)d * 3 + 0] - pos[(size_t)s * 3 + 0];
        sD3[wave][lane][1] = pos[(size_t)d * 3 + 1] - pos[(size_t)s * 3 + 1];
        sD3[wave][lane][2] = pos[(size_t)d * 3 + 2] - pos[(size_t)s * 3 + 2];
    }
    ds_fence();

    // pos-MLP hidden layer (K=3) computed by VALU directly into A-fragment layout
    float d3x = sD3[wave][m][0], d3y = sD3[wave][m][1], d3z = sD3[wave][m][2];
    bf16x16 p1f[2];
#pragma unroll
    for (int s = 0; s < 2; ++s) {
        bf16x16 f;
#pragma unroll
        for (int i = 0; i < 16; ++i) {
            int j = s * 32 + 8 * hh + (i & 7) + ((i >> 3) << 4);
            float val = d3x * sWp1[j * 4 + 0] + d3y * sWp1[j * 4 + 1] +
                        d3z * sWp1[j * 4 + 2] + sbp1[j];
            f[i] = (__bf16)fmaxf(val, 0.f);
        }
        p1f[s] = f;
    }

    // delta = relu(p1 @ Wp2 + bp2): kept in D-layout regs, bf16 copy to tile
    float deltaD[4][8];
#pragma unroll
    for (int t = 0; t < 4; ++t) {
        f32x8 acc = {};
        acc = wmma_bf16(p1f[0], load_b_frag(sWp2, lane, t, 0), acc);
        acc = wmma_bf16(p1f[1], load_b_frag(sWp2, lane, t, 32), acc);
        float bi = sbp2[t * 16 + n0];
#pragma unroll
        for (int e = 0; e < 8; ++e) {
            float dv = fmaxf(acc[e] + bi, 0.f);
            deltaD[t][e] = dv;
            tile[(e + 8 * hh) * CDIM + t * 16 + n0] = (__bf16)dv;
        }
    }
    ds_fence();

    // q = a_dst[dst] - a_src[src] + delta, built in A-fragment layout
    int srcm = sSrc[wave][m], dstm = sDst[wave][m];
    const float* adp = a_dst + (size_t)dstm * CDIM;
    const float* asp = a_src + (size_t)srcm * CDIM;
    bf16x16 qf[2];
#pragma unroll
    for (int s = 0; s < 2; ++s) {
        bf16x16 f;
#pragma unroll
        for (int c = 0; c < 2; ++c) {
            int jb = s * 32 + 8 * hh + c * 16;
            float4 ad0 = *(const float4*)(adp + jb);
            float4 ad1 = *(const float4*)(adp + jb + 4);
            float4 as0 = *(const float4*)(asp + jb);
            float4 as1 = *(const float4*)(asp + jb + 4);
            float adv[8] = {ad0.x, ad0.y, ad0.z, ad0.w, ad1.x, ad1.y, ad1.z, ad1.w};
            float asv[8] = {as0.x, as0.y, as0.z, as0.w, as1.x, as1.y, as1.z, as1.w};
#pragma unroll
            for (int i2 = 0; i2 < 8; ++i2) {
                float q = adv[i2] - asv[i2] + (float)tile[m * CDIM + jb + i2];
                f[c * 8 + i2] = (__bf16)q;
            }
        }
        qf[s] = f;
    }

    // t1 = relu(q @ Wa1 + ba1) -> tile (overwrites delta copy; deltaD in regs)
#pragma unroll
    for (int t = 0; t < 4; ++t) {
        f32x8 acc = {};
        acc = wmma_bf16(qf[0], load_b_frag(sWa1, lane, t, 0), acc);
        acc = wmma_bf16(qf[1], load_b_frag(sWa1, lane, t, 32), acc);
        float bi = sba1[t * 16 + n0];
#pragma unroll
        for (int e = 0; e < 8; ++e)
            tile[(e + 8 * hh) * CDIM + t * 16 + n0] = (__bf16)fmaxf(acc[e] + bi, 0.f);
    }
    ds_fence();

    bf16x16 t1f0 = load_a_frag(tile, lane, 0);
    bf16x16 t1f1 = load_a_frag(tile, lane, 32);

    // logits = relu(t1 @ Wa2 + ba2), then segment-max / weighted aggregation
#pragma unroll
    for (int t = 0; t < 4; ++t) {
        f32x8 acc = {};
        acc = wmma_bf16(t1f0, load_b_frag(sWa2, lane, t, 0), acc);
        acc = wmma_bf16(t1f1, load_b_frag(sWa2, lane, t, 32), acc);
        int n = t * 16 + n0;
        float bi = sba2[n];
#pragma unroll
        for (int e = 0; e < 8; ++e) {
            int row = e + 8 * hh;
            if (e0 + row >= E) continue;
            float logit = fmaxf(acc[e] + bi, 0.f);
            int dd = sDst[wave][row];
            size_t off = (size_t)dd * CDIM + n;
            if (PASS == 1) {
                unsigned bits = __float_as_uint(logit);
                unsigned key = (bits & 0x80000000u) ? ~bits : (bits | 0x80000000u);
                atomicMax(&node_max[off], key);   // global_atomic_max_u32
            } else {
                unsigned u = node_max[off];
                float mx = __uint_as_float((u & 0x80000000u) ? (u & 0x7FFFFFFFu) : ~u);
                float w = __expf(logit - mx);
                int ss = sSrc[wave][row];
                float term = w * (vv[(size_t)ss * CDIM + n] + deltaD[t][e]);
                __hip_atomic_fetch_add(&denom[off], w, __ATOMIC_RELAXED,
                                       __HIP_MEMORY_SCOPE_AGENT);
                __hip_atomic_fetch_add(&accum[off], term, __ATOMIC_RELAXED,
                                       __HIP_MEMORY_SCOPE_AGENT);
            }
        }
    }
}

// ---------------------------------------------------------------------------
// Kernel 4: out = relu((accum / (denom + 1e-16)) @ W_out + b_out)
// ---------------------------------------------------------------------------
__global__ __launch_bounds__(256) void node_out(
    const float* __restrict__ accum, const float* __restrict__ denom,
    const __bf16* __restrict__ wbf, const float* __restrict__ b_out,
    float* __restrict__ out, int N) {
    __shared__ __align__(16) __bf16 sW[4096];
    __shared__ float sb[64];
    __shared__ __align__(16) __bf16 sTile[8][16 * CDIM];

    int tid = threadIdx.x;
    for (int i = tid; i < 4096; i += 256) sW[i] = wbf[7 * 4096 + i];
    if (tid < 64) sb[tid] = b_out[tid];
    __syncthreads();

    int wave = tid >> 5, lane = tid & 31;
    int r0 = (blockIdx.x * 8 + wave) * 16;
    if (r0 >= N) return;
    __bf16* tile = sTile[wave];
    int n0 = lane & 15, hh = lane >> 4;

#pragma unroll
    for (int i = 0; i < 8; ++i) {
        int e4 = (lane + i * 32) * 4;
        int row = e4 >> 6, col = e4 & 63;
        float4 av = make_float4(0.f, 0.f, 0.f, 0.f);
        float4 dv = make_float4(0.f, 0.f, 0.f, 0.f);
        if (r0 + row < N) {
            size_t g = (size_t)(r0 + row) * CDIM + col;
            av = *(const float4*)(accum + g);
            dv = *(const float4*)(denom + g);
        }
        tile[row * CDIM + col + 0] = (__bf16)(av.x / (dv.x + 1e-16f));
        tile[row * CDIM + col + 1] = (__bf16)(av.y / (dv.y + 1e-16f));
        tile[row * CDIM + col + 2] = (__bf16)(av.z / (dv.z + 1e-16f));
        tile[row * CDIM + col + 3] = (__bf16)(av.w / (dv.w + 1e-16f));
    }
    ds_fence();

    bf16x16 a0 = load_a_frag(tile, lane, 0);
    bf16x16 a1 = load_a_frag(tile, lane, 32);
#pragma unroll
    for (int t = 0; t < 4; ++t) {
        f32x8 acc = {};
        acc = wmma_bf16(a0, load_b_frag(sW, lane, t, 0), acc);
        acc = wmma_bf16(a1, load_b_frag(sW, lane, t, 32), acc);
        float bi = sb[t * 16 + n0];
#pragma unroll
        for (int e = 0; e < 8; ++e) {
            int row = e + 8 * hh;
            if (r0 + row < N)
                out[(size_t)(r0 + row) * CDIM + t * 16 + n0] = fmaxf(acc[e] + bi, 0.f);
        }
    }
}

// ---------------------------------------------------------------------------
extern "C" void kernel_launch(void* const* d_in, const int* in_sizes, int n_in,
                              void* d_out, int out_size, void* d_ws, size_t ws_size,
                              hipStream_t stream) {
    const float* x     = (const float*)d_in[0];
    const float* pos   = (const float*)d_in[1];
    const float* W_in  = (const float*)d_in[2];
    const float* b_in  = (const float*)d_in[3];
    const float* W_src = (const float*)d_in[4];
    const float* W_dst = (const float*)d_in[5];
    const float* W_lin = (const float*)d_in[6];
    const float* b_lin = (const float*)d_in[7];
    const float* Wp1   = (const float*)d_in[8];
    const float* bp1   = (const float*)d_in[9];
    const float* Wp2   = (const float*)d_in[10];
    const float* bp2   = (const float*)d_in[11];
    const float* Wa1   = (const float*)d_in[12];
    const float* ba1   = (const float*)d_in[13];
    const float* Wa2   = (const float*)d_in[14];
    const float* ba2   = (const float*)d_in[15];
    const float* W_out = (const float*)d_in[16];
    const float* b_out = (const float*)d_in[17];
    const int*   eidx  = (const int*)d_in[18];

    int N = in_sizes[0] / 64;
    int E = in_sizes[18] / 2;
    size_t NC = (size_t)N * CDIM;

    char* ws = (char*)d_ws;
    __bf16*  wbf    = (__bf16*)ws;                  // 8*4096 bf16 = 64 KB
    float*   a_src  = (float*)(ws + 65536);
    float*   a_dst  = a_src + NC;
    float*   vv     = a_dst + NC;
    unsigned* nmax  = (unsigned*)(vv + NC);
    float*   denom  = (float*)(nmax + NC);
    float*   accum  = denom + NC;

    const int* esrc = eidx;
    const int* edst = eidx + E;

    prep_weights<<<(8 * 4096 + 255) / 256, 256, 0, stream>>>(
        W_in, W_src, W_dst, W_lin, Wp2, Wa1, Wa2, W_out, wbf);
    init_zero<<<2048, 256, 0, stream>>>(nmax, 3 * NC);   // nmax, denom, accum contiguous

    node_transform<<<(N + 127) / 128, 256, 0, stream>>>(
        x, b_in, b_lin, wbf, a_src, a_dst, vv, N);

    edge_pass<1><<<(E + 127) / 128, 256, 0, stream>>>(
        pos, esrc, edst, a_src, a_dst, vv, Wp1, bp1, bp2, ba1, ba2, wbf,
        nmax, denom, accum, E);
    edge_pass<2><<<(E + 127) / 128, 256, 0, stream>>>(
        pos, esrc, edst, a_src, a_dst, vv, Wp1, bp1, bp2, ba1, ba2, wbf,
        nmax, denom, accum, E);

    node_out<<<(N + 127) / 128, 256, 0, stream>>>(
        accum, denom, wbf, b_out, (float*)d_out, N);
}
